// DQGSA_50646254354999
// MI455X (gfx1250) — compile-verified, matrix operand order checked by reading, not run on hardware
//
#include <hip/hip_runtime.h>
#include <hip/hip_bf16.h>
#include <math.h>
#include <stdint.h>

// Problem constants (from reference)
#define BSZ   1024
#define NPIX  100
#define CCH   256
#define C4    1024
#define NTOK  (BSZ * NPIX)

typedef __attribute__((ext_vector_type(16))) __bf16 bf16x16;
typedef __attribute__((ext_vector_type(8)))  float  floatx8;
typedef unsigned int u32x4 __attribute__((ext_vector_type(4)));
typedef int          i32x4 __attribute__((ext_vector_type(4)));
typedef int          i32x8 __attribute__((ext_vector_type(8)));

#if defined(__has_builtin)
#if __has_builtin(__builtin_amdgcn_tensor_load_to_lds) && __has_builtin(__builtin_amdgcn_s_wait_tensorcnt)
#define HAVE_TDM 1
#endif
#endif

union BF { unsigned u[8]; bf16x16 v; };
union FA { float f[8]; floatx8 v; };

__device__ __forceinline__ unsigned short f2bf(float x) {
    union { float f; unsigned u; } a; a.f = x;
    unsigned r = a.u + 0x7FFFu + ((a.u >> 16) & 1u);  // round-to-nearest-even
    return (unsigned short)(r >> 16);
}

// K offset inside a 16x32 bf16 A fragment (and B^T fragment), per ISA 7.12.2:
// VGPR v, lane-half h -> K = 16*(v>=4) + 8*h + 2*(v&3), elements K,K+1 packed.
__device__ __forceinline__ int kOff(int v, int half) {
    return ((v & 4) << 2) + (half << 3) + ((v & 3) << 1);
}

// ---------------------------------------------------------------------------
// Weight prep: bf16 conversion + layout so B^T rows are contiguous in K.
// ---------------------------------------------------------------------------
__global__ void prep_conv_w(const float* __restrict__ w, unsigned short* __restrict__ out) {
    int i = blockIdx.x * 256 + threadIdx.x;
    if (i >= 256 * 2304) return;
    int cout = i / 2304, rem = i % 2304;
    int kk = rem / 256, cin = rem % 256;             // K reordered as tap-major
    out[i] = f2bf(w[cout * 2304 + cin * 9 + kk]);
}

__global__ void prep_w1t(const float* __restrict__ w, unsigned short* __restrict__ out) {
    int i = blockIdx.x * 256 + threadIdx.x;
    if (i >= C4 * CCH) return;
    int n = i / CCH, k = i % CCH;                    // w1: (256,1024) -> w1T: (1024,256)
    out[i] = f2bf(w[k * C4 + n]);
}

__global__ void prep_w2t(const float* __restrict__ w, unsigned short* __restrict__ out) {
    int i = blockIdx.x * 256 + threadIdx.x;
    if (i >= CCH * C4) return;
    int n = i / C4, k = i % C4;                      // w2: (1024,256) -> w2T: (256,1024)
    out[i] = f2bf(w[k * CCH + n]);
}

// ---------------------------------------------------------------------------
// conv2 (3x3, 256->256) as implicit GEMM, one workgroup per batch element.
// M = pixels (padded 112 -> 7 tiles), N = cout (16 tiles), K = 9*256 = 2304.
// ---------------------------------------------------------------------------
__global__ __launch_bounds__(256) void conv2_kernel(const float* __restrict__ x1,
                                                    const unsigned short* __restrict__ wconv,
                                                    float* __restrict__ aconv) {
    __shared__ unsigned short xs[NPIX * CCH];        // 51.2 KB bf16 image (p, c)
    const int n   = blockIdx.x;
    const int tid = threadIdx.x;

    const float* xsrc = x1 + (size_t)n * NPIX * CCH;
    for (int i = tid; i < NPIX * CCH; i += 256) xs[i] = f2bf(xsrc[i]);
    __syncthreads();

    const int wave = tid >> 5;
    const int lane = tid & 31;
    const int col  = lane & 15;     // N column (and A row m)
    const int half = lane >> 4;
    const int nt0  = wave * 2;      // 8 waves x 2 N-tiles = 16 N-tiles (cout 256)

    float* outb = aconv + (size_t)n * NPIX * CCH;

    for (int mt = 0; mt < 7; ++mt) {
        FA acc0, acc1;
        const floatx8 z8 = {0.f, 0.f, 0.f, 0.f, 0.f, 0.f, 0.f, 0.f};
        acc0.v = z8; acc1.v = z8;

        const int p  = mt * 16 + col;     // A-row pixel for this lane
        const int py = p / 10, px = p % 10;
        const bool prow = (p < NPIX);

        for (int kk = 0; kk < 9; ++kk) {
            const int dy = kk / 3 - 1, dx = kk % 3 - 1;
            const int yy = py + dy, xx = px + dx;
            const bool valid = prow && (yy >= 0) && (yy < 10) && (xx >= 0) && (xx < 10);
            const int psrc = valid ? (yy * 10 + xx) : 0;   // clamp, zero later
            const unsigned short* arow  = xs + psrc * CCH;
            const unsigned short* brow0 = wconv + (size_t)(nt0 * 16 + col) * 2304 + kk * 256;
            const unsigned short* brow1 = brow0 + (size_t)16 * 2304;
            // prefetch next tap's weight rows into cache (global_prefetch_b8)
            __builtin_prefetch(brow0 + 256, 0, 1);
            __builtin_prefetch(brow1 + 256, 0, 1);

            for (int ks = 0; ks < 8; ++ks) {
                const int kb = ks * 32;
                BF af, bf0, bf1;
#pragma unroll
                for (int v = 0; v < 8; ++v) {
                    const int ko = kOff(v, half);
                    unsigned av = *(const unsigned*)(arow + kb + ko);
                    af.u[v]  = valid ? av : 0u;            // per-lane zero pad
                    bf0.u[v] = *(const unsigned*)(brow0 + kb + ko);
                    bf1.u[v] = *(const unsigned*)(brow1 + kb + ko);
                }
                acc0.v = __builtin_amdgcn_wmma_f32_16x16x32_bf16(false, af.v, false, bf0.v,
                                                                 (short)0, acc0.v, false, false);
                acc1.v = __builtin_amdgcn_wmma_f32_16x16x32_bf16(false, af.v, false, bf1.v,
                                                                 (short)0, acc1.v, false, false);
            }
        }
        // C/D layout: VGPR r -> M = r + 8*half, N = lane&15
#pragma unroll
        for (int r = 0; r < 8; ++r) {
            const int pr = mt * 16 + r + 8 * half;
            if (pr < NPIX) {
                outb[pr * CCH + nt0 * 16 + col]       = acc0.f[r];
                outb[pr * CCH + (nt0 + 1) * 16 + col] = acc1.f[r];
            }
        }
    }
}

// ---------------------------------------------------------------------------
// Distance gating + CBAM spatial attention + LayerNorm (bandwidth-bound),
// one workgroup per batch element, pixel-per-thread. Emits bf16 LN tokens.
// ---------------------------------------------------------------------------
__global__ __launch_bounds__(128) void gate_ln_kernel(const float* __restrict__ x2,
                                                      const float* __restrict__ aconv,
                                                      const float* __restrict__ conv3w,
                                                      const float* __restrict__ conv1w,
                                                      const float* __restrict__ lnw,
                                                      const float* __restrict__ lnb,
                                                      unsigned short* __restrict__ xln) {
    __shared__ float dist[NPIX], wgs[NPIX], avgs[NPIX], maxs[NPIX], sfs[NPIX];
    const int n = blockIdx.x, tid = threadIdx.x;
    const float* A = aconv + (size_t)n * NPIX * CCH;
    const float* B = x2    + (size_t)n * NPIX * CCH;

    if (tid < NPIX) {                                 // per-pixel L2 distance
        float ssd = 0.f;
        for (int c = 0; c < CCH; ++c) { float d = B[tid * CCH + c] - A[tid * CCH + c]; ssd += d * d; }
        dist[tid] = sqrtf(ssd) + 1e-8f;
    }
    __syncthreads();
    if (tid < NPIX) {                                 // 3x3 conv on dist, sigmoid(0.5*)
        const int y = tid / 10, x = tid % 10; float acc = 0.f;
        for (int dy = -1; dy <= 1; ++dy)
            for (int dx = -1; dx <= 1; ++dx) {
                const int yy = y + dy, xx = x + dx;
                if (yy >= 0 && yy < 10 && xx >= 0 && xx < 10)
                    acc += conv3w[(dy + 1) * 3 + (dx + 1)] * dist[yy * 10 + xx];
            }
        wgs[tid] = 1.f / (1.f + expf(-acc * 0.5f));
    }
    __syncthreads();
    if (tid < NPIX) {                                 // channel mean/max of mix
        const float w = wgs[tid]; float s = 0.f, mx = -3.4e38f;
        for (int c = 0; c < CCH; ++c) {
            const float mix = w * A[tid * CCH + c] + (1.f - w) * B[tid * CCH + c];
            s += mix; mx = fmaxf(mx, mix);
        }
        avgs[tid] = s * (1.f / CCH); maxs[tid] = mx;
    }
    __syncthreads();
    if (tid < NPIX) {                                 // 7x7 conv on [avg,max], sigmoid
        const int y = tid / 10, x = tid % 10; float acc = 0.f;
        for (int dy = -3; dy <= 3; ++dy)
            for (int dx = -3; dx <= 3; ++dx) {
                const int yy = y + dy, xx = x + dx;
                if (yy >= 0 && yy < 10 && xx >= 0 && xx < 10) {
                    const int q = yy * 10 + xx, wi = (dy + 3) * 7 + (dx + 3);
                    acc += conv1w[wi] * avgs[q] + conv1w[49 + wi] * maxs[q];
                }
            }
        sfs[tid] = 1.f / (1.f + expf(-acc));
    }
    __syncthreads();
    if (tid < NPIX) {                                 // x = s*a; LayerNorm over C
        const float s = sfs[tid]; float sum = 0.f, sq = 0.f;
        for (int c = 0; c < CCH; ++c) { const float v = s * A[tid * CCH + c]; sum += v; sq += v * v; }
        const float mu  = sum * (1.f / CCH);
        const float var = sq * (1.f / CCH) - mu * mu;
        const float rinv = rsqrtf(var + 1e-6f);
        unsigned short* orow = xln + ((size_t)n * NPIX + tid) * CCH;
        for (int c = 0; c < CCH; ++c) {
            const float v = (s * A[tid * CCH + c] - mu) * rinv * lnw[c] + lnb[c];
            orow[c] = f2bf(v);
        }
    }
}

// ---------------------------------------------------------------------------
// Fused MLP: 16-token tile -> GEMM1(256->1024) + gelu -> LDS -> GEMM2(1024->256)
// -> * gamma + residual(x2). h never touches HBM. A-tile staged via the
// Tensor Data Mover (tensor_load_to_lds, TENSORcnt) when available.
// ---------------------------------------------------------------------------
__global__ __launch_bounds__(256) void mlp_kernel(const unsigned short* __restrict__ xln,
                                                  const unsigned short* __restrict__ w1t,
                                                  const float* __restrict__ b1,
                                                  const unsigned short* __restrict__ w2t,
                                                  const float* __restrict__ b2,
                                                  const float* __restrict__ gamma,
                                                  const float* __restrict__ x2,
                                                  float* __restrict__ out) {
    __shared__ unsigned short As[16 * CCH];           // 8 KB
    __shared__ unsigned short Hs[16 * C4];            // 32 KB
    const int tok0 = blockIdx.x * 16;
    const int tid  = threadIdx.x;

#if defined(HAVE_TDM)
    if ((tid >> 5) == 0) {  // one TDM issue per block (EXEC ignored, wave-level op)
        // D# group0: count=1 | lds_addr | global_addr | type=2 ("image")
        const unsigned long long ga = (unsigned long long)(uintptr_t)(xln + (size_t)tok0 * CCH);
        u32x4 g0;
        g0[0] = 1u;                                         // count=1, user mode
        g0[1] = (unsigned)(uintptr_t)(void*)As;             // LDS byte address
        g0[2] = (unsigned)(ga & 0xFFFFFFFFull);             // global_addr[31:0]
        g0[3] = (unsigned)((ga >> 32) & 0x1FFFFFFull) | (2u << 30);  // [56:32] | type=2
        // D# group1: 1-D copy of 4096 bf16 elements (data_size=2B)
        const int NE = 16 * CCH;                            // 4096 elements
        i32x8 g1;
        g1[0] = 1 << 16;                                    // data_size=1 -> 2 bytes
        g1[1] = (NE & 0xFFFF) << 16;                        // tensor_dim0[15:0]
        g1[2] = ((NE >> 16) & 0xFFFF) | (1 << 16);          // tensor_dim0[31:16] | tensor_dim1=1
        g1[3] = (NE & 0xFFFF) << 16;                        // tile_dim0 = 4096
        g1[4] = 1;                                          // tile_dim1 = 1
        g1[5] = NE;                                         // tensor_dim0_stride[31:0]
        g1[6] = 0; g1[7] = 0;
        const i32x4 gz4 = {0, 0, 0, 0};
#if __clang_major__ >= 23
        const i32x8 gz8 = {0, 0, 0, 0, 0, 0, 0, 0};
        __builtin_amdgcn_tensor_load_to_lds(g0, g1, gz4, gz4, gz8, 0);
#else
        __builtin_amdgcn_tensor_load_to_lds(g0, g1, gz4, gz4, 0);
#endif
        __builtin_amdgcn_s_wait_tensorcnt(0);               // drain TDM before barrier
    }
#else
    {   // fallback: manual staging of 16 bf16 tokens into LDS
        const unsigned* src = (const unsigned*)(xln + (size_t)tok0 * CCH);
        unsigned* dst = (unsigned*)As;
        for (int i = tid; i < 16 * CCH / 2; i += 256) dst[i] = src[i];
    }
#endif
    __syncthreads();

    const int wave = tid >> 5, lane = tid & 31, col = lane & 15, half = lane >> 4;
    const floatx8 z8 = {0.f, 0.f, 0.f, 0.f, 0.f, 0.f, 0.f, 0.f};

    // --- GEMM1: (16 x 256) @ (256 x 1024); wave owns 8 N-tiles ---
    FA acc1[8];
#pragma unroll
    for (int j = 0; j < 8; ++j) acc1[j].v = z8;
    const unsigned short* arow = As + col * CCH;
    for (int ks = 0; ks < 8; ++ks) {
        const int kb = ks * 32;
        BF af;
#pragma unroll
        for (int v = 0; v < 8; ++v) af.u[v] = *(const unsigned*)(arow + kb + kOff(v, half));
#pragma unroll
        for (int j = 0; j < 8; ++j) {
            const int ng = (wave * 8 + j) * 16 + col;
            const unsigned short* brow = w1t + (size_t)ng * CCH + kb;
            __builtin_prefetch(brow + 32, 0, 1);            // next K-step of this row
            BF bf;
#pragma unroll
            for (int v = 0; v < 8; ++v) bf.u[v] = *(const unsigned*)(brow + kOff(v, half));
            acc1[j].v = __builtin_amdgcn_wmma_f32_16x16x32_bf16(false, af.v, false, bf.v,
                                                                (short)0, acc1[j].v, false, false);
        }
    }
    // bias + exact gelu -> bf16 h tile in LDS
#pragma unroll
    for (int j = 0; j < 8; ++j) {
        const int ng = (wave * 8 + j) * 16 + col;
        const float bias = b1[ng];
#pragma unroll
        for (int r = 0; r < 8; ++r) {
            const float v = acc1[j].f[r] + bias;
            const float g = 0.5f * v * (1.f + erff(v * 0.70710678118f));
            Hs[(r + 8 * half) * C4 + ng] = f2bf(g);
        }
    }
    __syncthreads();

    // --- GEMM2: (16 x 1024) @ (1024 x 256); wave owns 2 N-tiles ---
    FA acc2[2];
    acc2[0].v = z8; acc2[1].v = z8;
    const unsigned short* hrow = Hs + col * C4;
    for (int ks = 0; ks < 32; ++ks) {
        const int kb = ks * 32;
        BF af;
#pragma unroll
        for (int v = 0; v < 8; ++v) af.u[v] = *(const unsigned*)(hrow + kb + kOff(v, half));
#pragma unroll
        for (int t = 0; t < 2; ++t) {
            const int ng = (wave * 2 + t) * 16 + col;
            const unsigned short* brow = w2t + (size_t)ng * C4 + kb;
            __builtin_prefetch(brow + 32, 0, 1);
            BF bf;
#pragma unroll
            for (int v = 0; v < 8; ++v) bf.u[v] = *(const unsigned*)(brow + kOff(v, half));
            acc2[t].v = __builtin_amdgcn_wmma_f32_16x16x32_bf16(false, af.v, false, bf.v,
                                                                (short)0, acc2[t].v, false, false);
        }
    }
    // * gamma + b2 + residual x2  (final output is token-major == x2 layout)
#pragma unroll
    for (int t = 0; t < 2; ++t) {
        const int cch = (wave * 2 + t) * 16 + col;
        const float bias = b2[cch], gm = gamma[cch];
#pragma unroll
        for (int r = 0; r < 8; ++r) {
            const int tok = tok0 + r + 8 * half;
            const size_t o = (size_t)tok * CCH + cch;
            out[o] = (acc2[t].f[r] + bias) * gm + x2[o];
        }
    }
}

// ---------------------------------------------------------------------------
extern "C" void kernel_launch(void* const* d_in, const int* in_sizes, int n_in,
                              void* d_out, int out_size, void* d_ws, size_t ws_size,
                              hipStream_t stream) {
    const float* x1     = (const float*)d_in[0];
    const float* x2     = (const float*)d_in[1];
    const float* conv2w = (const float*)d_in[2];
    const float* conv3w = (const float*)d_in[3];
    const float* conv1w = (const float*)d_in[4];
    const float* lnw    = (const float*)d_in[5];
    const float* lnb    = (const float*)d_in[6];
    const float* w1     = (const float*)d_in[7];
    const float* b1     = (const float*)d_in[8];
    const float* w2     = (const float*)d_in[9];
    const float* b2     = (const float*)d_in[10];
    const float* gm     = (const float*)d_in[11];
    float* out = (float*)d_out;

    char* ws = (char*)d_ws;
    unsigned short* wconv = (unsigned short*)(ws + 0);                 // 1,179,648 B
    unsigned short* w1t   = (unsigned short*)(ws + 1179648);           //   524,288 B
    unsigned short* w2t   = (unsigned short*)(ws + 1703936);           //   524,288 B
    float*          aconv = (float*)(ws + 2228224);                    // 104,857,600 B
    unsigned short* xln   = (unsigned short*)(ws + 2228224 + (size_t)NTOK * CCH * 4);

    prep_conv_w<<<(256 * 2304 + 255) / 256, 256, 0, stream>>>(conv2w, wconv);
    prep_w1t<<<(C4 * CCH + 255) / 256, 256, 0, stream>>>(w1, w1t);
    prep_w2t<<<(CCH * C4 + 255) / 256, 256, 0, stream>>>(w2, w2t);

    conv2_kernel<<<BSZ, 256, 0, stream>>>(x1, wconv, aconv);
    gate_ln_kernel<<<BSZ, 128, 0, stream>>>(x2, aconv, conv3w, conv1w, lnw, lnb, xln);
    mlp_kernel<<<NTOK / 16, 256, 0, stream>>>(xln, w1t, b1, w2t, b2, gm, x2, out);
}
